// Coteachingloss_dropregionce_52312701665506
// MI455X (gfx1250) — compile-verified, matrix operand order checked by reading, not run on hardware
//
#include <hip/hip_runtime.h>
#include <hip/hip_bf16.h>
#include <cstdint>
#include <cstddef>

// Problem constants (match reference)
#define Bn   8
#define Cn   19
#define Hn   512
#define Wn   512
#define HPn  256
#define WPn  256
#define Ln   (HPn * WPn)      // 65536 pooled pixels per batch row
#define NLOSS (Bn * Ln)       // 524288

typedef __attribute__((ext_vector_type(2)))  float    f2;
typedef __attribute__((ext_vector_type(4)))  float    f4;
typedef __attribute__((ext_vector_type(4)))  int      i4;
typedef __attribute__((ext_vector_type(2)))  float    v2f;
typedef __attribute__((ext_vector_type(8)))  float    v8f;
typedef __attribute__((ext_vector_type(16))) _Float16 v16h;

// ---------------- workspace layout (bytes) ----------------
static constexpr size_t OFF_L1   = 0;                               // loss1 [8][65536] f32
static constexpr size_t OFF_L2   = OFF_L1 + (size_t)NLOSS * 4;      // loss2 [8][65536] f32
static constexpr size_t OFF_H1   = OFF_L2 + (size_t)NLOSS * 4;      // hist pass1: 16 slices x 4096 ints
static constexpr size_t OFF_H2   = OFF_H1 + (size_t)16 * 4096 * 4;  // hist pass2
static constexpr size_t OFF_ST   = OFF_H2 + (size_t)16 * 4096 * 4;  // state: 16 slices x 8 ints
static constexpr size_t OFF_PART = OFF_ST + (size_t)16 * 8 * 4;     // partials: 16 blk x 8 rows x 2 f32
static constexpr size_t OFF_END  = OFF_PART + (size_t)16 * 8 * 2 * 4;

__device__ __forceinline__ int calc_nr(const float* fr) {
    // matches python int((1.0 - forget_rate) * L): truncation toward zero
    return (int)((1.0f - fr[0]) * (float)Ln);
}

// Online log-sum-exp update: keeps per-pixel state in 2 VGPRs (no arrays -> no spill)
__device__ __forceinline__ void lse_update(float& m, float& s, float v) {
    float mn = fmaxf(m, v);
    s = s * __expf(m - mn) + __expf(v - mn);
    m = mn;
}

// ---------------- WMMA 64-value fp32 sum step ----------------
// A (16x4 f32) holds 64 arbitrary addends (2 per lane); B's rows each sum to 1
// (ones placed on lanes with lane%16==0), so sum over ALL entries of
// D = A*B + C equals sum(A) + sum(C) regardless of exact slot mapping.
__device__ __forceinline__ v8f wmma_sum64(float x0, float x1, float bval, v8f acc) {
#if __has_builtin(__builtin_amdgcn_wmma_f32_16x16x4_f32)
    v2f a; a.x = x0;   a.y = x1;
    v2f b; b.x = bval; b.y = bval;
    return __builtin_amdgcn_wmma_f32_16x16x4_f32(false, a, false, b, (short)0, acc,
                                                 false, false);
#else
    v16h a, b;
#pragma unroll
    for (int i = 0; i < 16; ++i) { a[i] = (_Float16)0.0f; b[i] = (_Float16)bval; }
    a[0] = (_Float16)x0; a[1] = (_Float16)x1;
    return __builtin_amdgcn_wmma_f32_16x16x32_f16(false, a, false, b, (short)0, acc,
                                                  false, false);
#endif
}

// ---------------- kernel 0: zero scratch ----------------
__global__ __launch_bounds__(256) void zero_kernel(int* __restrict__ p, int n) {
    int i = blockIdx.x * 256 + threadIdx.x;
    if (i < n) p[i] = 0;
}

// ---------------- kernel 1: fused 2x2 maxpool + CE (streaming, b128 loads) --------
// One thread produces TWO horizontally-adjacent pooled pixels, so every global
// load is a 16-byte (b128) nontemporal load: 512 B per wave-instruction.
__global__ __launch_bounds__(256) void pool_ce_kernel(const float* __restrict__ in1,
                                                      const float* __restrict__ in2,
                                                      const int*   __restrict__ tgt,
                                                      float* __restrict__ loss1,
                                                      float* __restrict__ loss2) {
    const int tid = blockIdx.x * 256 + threadIdx.x;   // 0 .. NLOSS/2 - 1 exactly
    const int b  = tid >> 15;                         // / 32768 pairs per row
    const int r  = tid & 32767;
    const int y  = r >> 7;                            // pooled row 0..255
    const int xp = r & 127;                           // pixel-pair index 0..127
    const int row0 = 2 * y, col0 = 4 * xp;

    // pooled targets for the two pixels (max of 4 ints each)
    const int* tb = tgt + ((size_t)(b * Hn + row0) * Wn + col0);
    i4 tr0 = __builtin_nontemporal_load((const i4*)tb);
    i4 tr1 = __builtin_nontemporal_load((const i4*)(tb + Wn));
    int tA = max(max(tr0.x, tr0.y), max(tr1.x, tr1.y));
    int tB = max(max(tr0.z, tr0.w), max(tr1.z, tr1.w));
    const bool validA = (tA != 255), validB = (tB != 255);
    tA = tA < 0 ? 0 : (tA > Cn - 1 ? Cn - 1 : tA);
    tB = tB < 0 ? 0 : (tB > Cn - 1 ? Cn - 1 : tB);

    const size_t cs = (size_t)Hn * Wn;                // channel stride
    size_t off = ((size_t)(b * Cn) * Hn + row0) * Wn + col0;

    // online LSE state: {input1,input2} x {pixelA,pixelB}
    float m1A = -3.4e38f, s1A = 0.0f, pt1A = 0.0f;
    float m1B = -3.4e38f, s1B = 0.0f, pt1B = 0.0f;
    float m2A = -3.4e38f, s2A = 0.0f, pt2A = 0.0f;
    float m2B = -3.4e38f, s2B = 0.0f, pt2B = 0.0f;

#pragma unroll 1
    for (int c = 0; c < Cn; ++c) {
        __builtin_prefetch(in1 + off + cs, 0, 0);     // global_prefetch_b8
        __builtin_prefetch(in2 + off + cs, 0, 0);
        f4 a0 = __builtin_nontemporal_load((const f4*)(in1 + off));
        f4 a1 = __builtin_nontemporal_load((const f4*)(in1 + off + Wn));
        f4 b0 = __builtin_nontemporal_load((const f4*)(in2 + off));
        f4 b1 = __builtin_nontemporal_load((const f4*)(in2 + off + Wn));
        float v1A = fmaxf(fmaxf(a0.x, a0.y), fmaxf(a1.x, a1.y));
        float v1B = fmaxf(fmaxf(a0.z, a0.w), fmaxf(a1.z, a1.w));
        float v2A = fmaxf(fmaxf(b0.x, b0.y), fmaxf(b1.x, b1.y));
        float v2B = fmaxf(fmaxf(b0.z, b0.w), fmaxf(b1.z, b1.w));
        lse_update(m1A, s1A, v1A);
        lse_update(m1B, s1B, v1B);
        lse_update(m2A, s2A, v2A);
        lse_update(m2B, s2B, v2B);
        pt1A = (c == tA) ? v1A : pt1A;
        pt1B = (c == tB) ? v1B : pt1B;
        pt2A = (c == tA) ? v2A : pt2A;
        pt2B = (c == tB) ? v2B : pt2B;
        off += cs;
    }

    f2 o1, o2;
    o1.x = validA ? (m1A + __logf(s1A) - pt1A) : 0.0f;
    o1.y = validB ? (m1B + __logf(s1B) - pt1B) : 0.0f;
    o2.x = validA ? (m2A + __logf(s2A) - pt2A) : 0.0f;
    o2.y = validB ? (m2B + __logf(s2B) - pt2B) : 0.0f;
    const size_t o = (size_t)b * Ln + (size_t)y * WPn + 2 * xp;
    *(f2*)(loss1 + o) = o1;
    *(f2*)(loss2 + o) = o2;
}

// ---------------- selection: radix histograms on fp32 bits (nonneg => monotone) ----
// slice s in [0,16): array a = s>>3 (0=loss1, 1=loss2), row b = s&7

__global__ __launch_bounds__(256) void hist1_kernel(const float* __restrict__ loss1,
                                                    const float* __restrict__ loss2,
                                                    int* __restrict__ hist1) {
    const int slice = blockIdx.y;
    const float* src = ((slice < 8) ? loss1 : loss2) + (size_t)(slice & 7) * Ln;
    __shared__ int h[4096];
    for (int i = threadIdx.x; i < 4096; i += 256) h[i] = 0;
    __syncthreads();
    const int per = Ln / gridDim.x;
    const int start = blockIdx.x * per;
    for (int i = threadIdx.x; i < per; i += 256) {
        unsigned k = __float_as_uint(src[start + i]);
        atomicAdd(&h[k >> 19], 1);
    }
    __syncthreads();
    int* gh = hist1 + slice * 4096;
    for (int i = threadIdx.x; i < 4096; i += 256) {
        int v = h[i];
        if (v) atomicAdd(&gh[i], v);
    }
}

__global__ __launch_bounds__(256) void scan1_kernel(const int* __restrict__ hist1,
                                                    int* __restrict__ state,
                                                    const float* __restrict__ fr) {
    const int slice = blockIdx.x;
    const int* gh = hist1 + slice * 4096;
    __shared__ int seg[256];
    const int tid = threadIdx.x;
    int s = 0;
#pragma unroll
    for (int j = 0; j < 16; ++j) s += gh[tid * 16 + j];
    seg[tid] = s;
    __syncthreads();
    if (tid == 0) {
        const int NR = calc_nr(fr);
        int bin = 0, below = 0;
        if (NR > 0) {
            int cum = 0; bin = -1;
            for (int i = 0; i < 256 && bin < 0; ++i) {
                if (cum + seg[i] >= NR) {
                    int c = cum;
                    for (int j = 0; j < 16; ++j) {
                        int hv = gh[i * 16 + j];
                        if (c + hv >= NR) { bin = i * 16 + j; below = c; break; }
                        c += hv;
                    }
                } else cum += seg[i];
            }
            if (bin < 0) { bin = 4095; below = 0; }
        }
        int* st = state + slice * 8;
        st[0] = bin; st[1] = below;
    }
}

__global__ __launch_bounds__(256) void hist2_kernel(const float* __restrict__ loss1,
                                                    const float* __restrict__ loss2,
                                                    const int* __restrict__ state,
                                                    int* __restrict__ hist2) {
    const int slice = blockIdx.y;
    const int bin1 = state[slice * 8 + 0];
    const float* src = ((slice < 8) ? loss1 : loss2) + (size_t)(slice & 7) * Ln;
    __shared__ int h[4096];
    for (int i = threadIdx.x; i < 4096; i += 256) h[i] = 0;
    __syncthreads();
    const int per = Ln / gridDim.x;
    const int start = blockIdx.x * per;
    for (int i = threadIdx.x; i < per; i += 256) {
        unsigned k = __float_as_uint(src[start + i]);
        if ((int)(k >> 19) == bin1) atomicAdd(&h[(k >> 7) & 0xFFF], 1);
    }
    __syncthreads();
    int* gh = hist2 + slice * 4096;
    for (int i = threadIdx.x; i < 4096; i += 256) {
        int v = h[i];
        if (v) atomicAdd(&gh[i], v);
    }
}

__global__ __launch_bounds__(256) void scan2_kernel(const int* __restrict__ hist2,
                                                    int* __restrict__ state,
                                                    const float* __restrict__ fr) {
    const int slice = blockIdx.x;
    const int* gh = hist2 + slice * 4096;
    __shared__ int seg[256];
    const int tid = threadIdx.x;
    int s = 0;
#pragma unroll
    for (int j = 0; j < 16; ++j) s += gh[tid * 16 + j];
    seg[tid] = s;
    __syncthreads();
    if (tid == 0) {
        int* st = state + slice * 8;
        const int NR = calc_nr(fr);
        const int bin1 = st[0];
        const int below1 = st[1];
        int bin2 = 0, below2 = 0, band = 0;
        if (NR > 0) {
            const int need = NR - below1;             // elements needed inside bin1
            int cum = 0; bin2 = -1;
            for (int i = 0; i < 256 && bin2 < 0; ++i) {
                if (cum + seg[i] >= need) {
                    int c = cum;
                    for (int j = 0; j < 16; ++j) {
                        int hv = gh[i * 16 + j];
                        if (c + hv >= need) { bin2 = i * 16 + j; below2 = c; break; }
                        c += hv;
                    }
                } else cum += seg[i];
            }
            if (bin2 < 0) { bin2 = 4095; below2 = 0; }
            band = gh[bin2];
        }
        st[2] = (bin1 << 12) | bin2;                  // 24-bit prefix of key>>7
        st[3] = below1 + below2;                      // strictly-below count
        st[4] = band;                                 // band population
    }
}

// ---------------- kernel 7: masked co-teaching sums via WMMA ----------------
__global__ __launch_bounds__(256) void wmma_sum_kernel(const float* __restrict__ loss1,
                                                       const float* __restrict__ loss2,
                                                       const int* __restrict__ state,
                                                       const float* __restrict__ fr,
                                                       float* __restrict__ part) {
    const int b = blockIdx.y;
    const int NR = calc_nr(fr);
    const int* st1 = state + b * 8;        // thresholds of loss1 (ind1 -> weights loss2)
    const int* st2 = state + (8 + b) * 8;  // thresholds of loss2 (ind2 -> weights loss1)
    int p1 = st1[2]; const int below1 = st1[3]; const int band1 = st1[4];
    int p2 = st2[2]; const int below2 = st2[3]; const int band2 = st2[4];
    float w1 = (band1 > 0) ? (float)(NR - below1) / (float)band1 : 0.0f;
    float w2 = (band2 > 0) ? (float)(NR - below2) / (float)band2 : 0.0f;
    if (NR <= 0) { p1 = -1; p2 = -1; w1 = 0.0f; w2 = 0.0f; }

    const float* L1 = loss1 + (size_t)b * Ln;
    const float* L2 = loss2 + (size_t)b * Ln;

    const int wave = threadIdx.x >> 5;
    const int lane = threadIdx.x & 31;
    const int wavesPerRow = gridDim.x * 8;
    const int waveId = blockIdx.x * 8 + wave;
    const float bval = ((lane & 15) == 0) ? 1.0f : 0.0f;

    v8f acc1 = {0.f, 0.f, 0.f, 0.f, 0.f, 0.f, 0.f, 0.f};
    v8f acc2 = {0.f, 0.f, 0.f, 0.f, 0.f, 0.f, 0.f, 0.f};

    const int chunks = Ln / 64;            // 1024, exact
    for (int ch = waveId; ch < chunks; ch += wavesPerRow) {
        const int idx = ch * 64 + lane * 2;
        f2 xv = *(const f2*)(L1 + idx);
        f2 yv = *(const f2*)(L2 + idx);
        int k1a = (int)(__float_as_uint(xv.x) >> 7);
        int k1b = (int)(__float_as_uint(xv.y) >> 7);
        int k2a = (int)(__float_as_uint(yv.x) >> 7);
        int k2b = (int)(__float_as_uint(yv.y) >> 7);
        // loss1_update selected by loss2's ranking; loss2_update by loss1's
        float s2a = (k2a < p2) ? 1.0f : ((k2a == p2) ? w2 : 0.0f);
        float s2b = (k2b < p2) ? 1.0f : ((k2b == p2) ? w2 : 0.0f);
        float s1a = (k1a < p1) ? 1.0f : ((k1a == p1) ? w1 : 0.0f);
        float s1b = (k1b < p1) ? 1.0f : ((k1b == p1) ? w1 : 0.0f);
        acc1 = wmma_sum64(xv.x * s2a, xv.y * s2b, bval, acc1);
        acc2 = wmma_sum64(yv.x * s1a, yv.y * s1b, bval, acc2);
    }

    // reduce all D entries (non-zero only in effective column 0)
    float r1 = 0.0f, r2 = 0.0f;
#pragma unroll
    for (int i = 0; i < 8; ++i) { r1 += acc1[i]; r2 += acc2[i]; }
#pragma unroll
    for (int off = 16; off > 0; off >>= 1) {
        r1 += __shfl_xor(r1, off, 32);
        r2 += __shfl_xor(r2, off, 32);
    }
    __shared__ float ps1[8], ps2[8];
    if (lane == 0) { ps1[wave] = r1; ps2[wave] = r2; }
    __syncthreads();
    if (threadIdx.x == 0) {
        float t1 = 0.0f, t2 = 0.0f;
        for (int i = 0; i < 8; ++i) { t1 += ps1[i]; t2 += ps2[i]; }   // fixed order
        float* p = part + (size_t)(b * gridDim.x + blockIdx.x) * 2;
        p[0] = t1; p[1] = t2;
    }
}

// ---------------- kernel 8: finalize means ----------------
__global__ __launch_bounds__(32) void finalize_kernel(const float* __restrict__ part,
                                                      const float* __restrict__ fr,
                                                      float* __restrict__ out, int nPart) {
    if (threadIdx.x == 0 && blockIdx.x == 0) {
        const int NR = calc_nr(fr);
        float s1 = 0.0f, s2 = 0.0f;
        for (int i = 0; i < nPart; ++i) { s1 += part[2 * i]; s2 += part[2 * i + 1]; }
        const float denom = (NR > 0) ? (float)Bn * (float)NR : 1.0f;
        out[0] = s1 / denom;
        out[1] = s2 / denom;
    }
}

// ---------------- host entry ----------------
extern "C" void kernel_launch(void* const* d_in, const int* in_sizes, int n_in,
                              void* d_out, int out_size, void* d_ws, size_t ws_size,
                              hipStream_t stream) {
    const float* in1 = (const float*)d_in[0];
    const float* in2 = (const float*)d_in[1];
    const int*   tgt = (const int*)d_in[2];
    const float* fr  = (const float*)d_in[3];
    float*       out = (float*)d_out;

    char* ws = (char*)d_ws;
    float* loss1 = (float*)(ws + OFF_L1);
    float* loss2 = (float*)(ws + OFF_L2);
    int*   hist1 = (int*)(ws + OFF_H1);
    int*   hist2 = (int*)(ws + OFF_H2);
    int*   state = (int*)(ws + OFF_ST);
    float* part  = (float*)(ws + OFF_PART);

    const int NB = 16;  // blocks per row in sum kernel

    const int nzero = (int)((OFF_END - OFF_H1) / 4);
    zero_kernel<<<(nzero + 255) / 256, 256, 0, stream>>>((int*)(ws + OFF_H1), nzero);

    pool_ce_kernel<<<(NLOSS / 2) / 256, 256, 0, stream>>>(in1, in2, tgt, loss1, loss2);

    hist1_kernel<<<dim3(64, 16), 256, 0, stream>>>(loss1, loss2, hist1);
    scan1_kernel<<<16, 256, 0, stream>>>(hist1, state, fr);
    hist2_kernel<<<dim3(64, 16), 256, 0, stream>>>(loss1, loss2, state, hist2);
    scan2_kernel<<<16, 256, 0, stream>>>(hist2, state, fr);

    wmma_sum_kernel<<<dim3(NB, Bn), 256, 0, stream>>>(loss1, loss2, state, fr, part);
    finalize_kernel<<<1, 32, 0, stream>>>(part, fr, out, Bn * NB);
}
